// Net2to2_69372311765325
// MI455X (gfx1250) — compile-verified, matrix operand order checked by reading, not run on hardware
//
#include <hip/hip_runtime.h>
#include <hip/hip_bf16.h>

#define DEV __device__ __forceinline__

typedef __attribute__((ext_vector_type(16))) __bf16 v16bf;
typedef __attribute__((ext_vector_type(8)))  __bf16 v8bf;
typedef __attribute__((ext_vector_type(8)))  float  v8f;

constexpr int Bn = 8, Nn = 128, Cc = 64, Ll = 3, Hh = 128;
constexpr int MP = Bn * Nn * Nn;          // 131072 pixels
constexpr float SLOPE = 0.01f;

// slope < 1  =>  leaky(x) == max(x, slope*x)   (2 VALU, no cmp/cndmask chain)
DEV float leakyf(float v) { return fmaxf(v, SLOPE * v); }

// Fragment gather from a row-major [row][k] array where the lane-held index
// (A: row m, B: col n) is lane&15 and k = 16*(v>>2) + 2*(v&3) + 8*(lane>>4) + e.
// Using the same k-map for A and B makes D-layout == B-layout element-for-element,
// so chained GEMMs need no LDS staging (pure in-lane register reuse).
DEV v16bf frag_rm(const __bf16* __restrict__ base, int ld, int koff, int lane) {
  const int m = lane & 15, hi = lane >> 4;
  v16bf f;
#pragma unroll
  for (int v = 0; v < 8; ++v) {
    const int k = koff + ((v >> 2) << 4) + ((v & 3) << 1) + (hi << 3);
    f[2 * v]     = base[m * ld + k];
    f[2 * v + 1] = base[m * ld + k + 1];
  }
  return f;
}

DEV v8f wmma_bf(v16bf a, v16bf b, v8f c) {
  return __builtin_amdgcn_wmma_f32_16x16x32_bf16(false, a, false, b,
                                                 (short)0, c, false, false);
}

DEV void load8f(const float* __restrict__ p, float* o) {
  const float4 a = *(const float4*)p;
  const float4 b = *(const float4*)(p + 4);
  o[0] = a.x; o[1] = a.y; o[2] = a.z; o[3] = a.w;
  o[4] = b.x; o[5] = b.y; o[6] = b.z; o[7] = b.w;
}

// ---- f32 -> bf16 convert --------------------------------------------------
__global__ void cvt_kernel(const float* __restrict__ x, __bf16* __restrict__ xb, int ntot) {
  for (int i = blockIdx.x * blockDim.x + threadIdx.x; i < ntot; i += gridDim.x * blockDim.x)
    xb[i] = (__bf16)x[i];
}

// ---- transpose + convert the two GEMM coef slices (b=9 identity, b=10 transpose)
__global__ void coefT_kernel(const float* __restrict__ coefs,
                             __bf16* __restrict__ c10t, __bf16* __restrict__ c11t) {
  const int idx = blockIdx.x * blockDim.x + threadIdx.x;   // [l][s][d]
  if (idx >= Ll * Cc * Cc) return;
  const int l = idx / (Cc * Cc), rem = idx % (Cc * Cc);
  const int s = rem / Cc, d = rem % Cc;
  const float* cf = coefs + ((size_t)l * Cc * Cc + (size_t)d * Cc + s) * 15;
  c10t[idx] = (__bf16)cf[9];
  c11t[idx] = (__bf16)cf[10];
}

// ---- MessageNet, pixel dim as N: H = leaky(W3 leaky(W2 leaky(W1 X^T)))*mask
// block = (n,i); 8 waves; wave owns 16 pixels (j-tile). No LDS, no barriers.
__global__ __launch_bounds__(256) void msgnet_kernel(
    const __bf16* __restrict__ xb, const float* __restrict__ mask,
    const __bf16* __restrict__ w1b, const float* __restrict__ b1,
    const __bf16* __restrict__ w2b, const float* __restrict__ b2,
    const __bf16* __restrict__ w3b, const float* __restrict__ b3,
    __bf16* __restrict__ hb) {
  const int tid = threadIdx.x, lane = tid & 31, wv = tid >> 5;
  const int n = blockIdx.x / Nn, i = blockIdx.x % Nn;
  const int j0 = wv * 16, cn = lane & 15, chi = lane >> 4;
  const size_t pxbase = (size_t)(n * Nn + i) * Nn + j0;

  // B-fragments of X^T (k = channel, n = pixel) straight from row-major x
  const __bf16* xrow = xb + pxbase * Cc;
  const v16bf bx0 = frag_rm(xrow, Cc, 0, lane);
  const v16bf bx1 = frag_rm(xrow, Cc, 32, lane);
  const float mk = mask[pxbase + cn];

  // GEMM1: (128h x 64c) x (64c x 16px) -> D tiles feed B-frags directly
  v16bf by[4];
#pragma unroll
  for (int ks = 0; ks < 4; ++ks) {
#pragma unroll
    for (int half = 0; half < 2; ++half) {
      const int t = ks * 2 + half;
      v8f acc = {};
      acc = wmma_bf(frag_rm(w1b + t * 16 * Cc, Cc, 0, lane), bx0, acc);
      acc = wmma_bf(frag_rm(w1b + t * 16 * Cc, Cc, 32, lane), bx1, acc);
      float bb[8]; load8f(b1 + t * 16 + chi * 8, bb);
#pragma unroll
      for (int r = 0; r < 8; ++r)
        by[ks][half * 8 + r] = (__bf16)leakyf(acc[r] + bb[r]);
    }
  }

  // GEMM2: (128 x 128) x (128 x 16px)
  v16bf bz[4];
#pragma unroll
  for (int ks = 0; ks < 4; ++ks) {
#pragma unroll
    for (int half = 0; half < 2; ++half) {
      const int t = ks * 2 + half;
      v8f acc = {};
#pragma unroll
      for (int kk = 0; kk < 4; ++kk)
        acc = wmma_bf(frag_rm(w2b + t * 16 * Hh, Hh, kk * 32, lane), by[kk], acc);
      float bb[8]; load8f(b2 + t * 16 + chi * 8, bb);
#pragma unroll
      for (int r = 0; r < 8; ++r)
        bz[ks][half * 8 + r] = (__bf16)leakyf(acc[r] + bb[r]);
    }
  }

  // GEMM3: (64 x 128) x (128 x 16px) -> contiguous b128 store per tile
#pragma unroll
  for (int t = 0; t < 4; ++t) {
    v8f acc = {};
#pragma unroll
    for (int kk = 0; kk < 4; ++kk)
      acc = wmma_bf(frag_rm(w3b + t * 16 * Hh, Hh, kk * 32, lane), bz[kk], acc);
    float bb[8]; load8f(b3 + t * 16 + chi * 8, bb);
    v8bf o;
#pragma unroll
    for (int r = 0; r < 8; ++r)
      o[r] = (__bf16)(leakyf(acc[r] + bb[r]) * mk);
    *(v8bf*)(hb + (pxbase + cn) * Cc + t * 16 + chi * 8) = o;
  }
}

// ---- reductions: rowS[n,c,i]=sum_j h, colS[n,c,p]=sum_q h[q,p], diag ------
__global__ __launch_bounds__(64) void reduce_kernel(
    const __bf16* __restrict__ hb, float* __restrict__ rowS,
    float* __restrict__ colS, float* __restrict__ diagv) {
  const int n = blockIdx.x / Nn, p = blockIdx.x % Nn, c = threadIdx.x;
  float rs = 0.f, cs = 0.f;
  for (int q = 0; q < Nn; ++q) {
    rs += (float)hb[((size_t)(n * Nn + p) * Nn + q) * Cc + c];
    cs += (float)hb[((size_t)(n * Nn + q) * Nn + p) * Cc + c];
  }
  const float dg = (float)hb[((size_t)(n * Nn + p) * Nn + p) * Cc + c];
  const size_t o = ((size_t)n * Cc + c) * Nn + p;
  rowS[o] = rs; colS[o] = cs; diagv[o] = dg;
}

__global__ __launch_bounds__(64) void reduce2_kernel(
    const float* __restrict__ diagv, const float* __restrict__ rowS,
    float* __restrict__ sumdiag, float* __restrict__ sumall) {
  const int n = blockIdx.x, c = threadIdx.x;
  float sd = 0.f, sa = 0.f;
  for (int i = 0; i < Nn; ++i) {
    const size_t o = ((size_t)n * Cc + c) * Nn + i;
    sd += diagv[o]; sa += rowS[o];
  }
  sumdiag[n * Cc + c] = sd; sumall[n * Cc + c] = sa;
}

// ---- per-batch constants: G (always), E (diag-only) -----------------------
__global__ __launch_bounds__(64) void ge_kernel(
    const float* __restrict__ cf, const float* __restrict__ sumdiag,
    const float* __restrict__ sumall, float* __restrict__ G, float* __restrict__ E) {
  const int n = blockIdx.x, s = threadIdx.x;
  float g = 0.f, e = 0.f;
  for (int d = 0; d < Cc; ++d) {
    const float sd = sumdiag[n * Cc + d], sa = sumall[n * Cc + d];
    const float* cc = cf + ((size_t)d * Cc + s) * 15;
    g += cc[13] * sd + cc[14] * sa;                // ops 14,15
    e += cc[1]  * sd + cc[4]  * sa;                // ops 2,5 (diag only)
  }
  G[n * Cc + s] = g; E[n * Cc + s] = e;
}

// ---- PGB[n,i,s] = P + G + bias (always);  DED = D + E + diag_bias (diag) --
__global__ __launch_bounds__(64) void rowvec_kernel(
    const float* __restrict__ cf, const float* __restrict__ rowS,
    const float* __restrict__ colS, const float* __restrict__ diagv,
    const float* __restrict__ G, const float* __restrict__ E,
    const float* __restrict__ bias, const float* __restrict__ dbias,
    float* __restrict__ PGB, float* __restrict__ Q, float* __restrict__ DED) {
  const int n = blockIdx.x / Nn, i = blockIdx.x % Nn, s = threadIdx.x;
  float p = 0.f, q = 0.f, dd = 0.f;
  for (int d = 0; d < Cc; ++d) {
    const size_t o = ((size_t)n * Cc + d) * Nn + i;
    const float rs = rowS[o], cs = colS[o], dg = diagv[o];
    const float* cc = cf + ((size_t)d * Cc + s) * 15;
    p  += cc[5] * cs + cc[6] * rs + cc[11] * dg;   // ops 6,7,12
    q  += cc[7] * cs + cc[8] * rs + cc[12] * dg;   // ops 8,9,13
    dd += cc[0] * dg + cc[2] * rs + cc[3]  * cs;   // ops 1,3,4 (diag only)
  }
  const size_t o = ((size_t)(n * Nn) + i) * Cc + s;
  PGB[o] = p + G[n * Cc + s] + bias[s];
  Q[o]   = q;
  DED[o] = dd + E[n * Cc + s] + dbias[s];
}

// ---- Eq2to2 output, pixel dim as N: O = C10^T h1 + C11^T h2 + broadcasts --
__global__ __launch_bounds__(256) void out_kernel(
    const __bf16* __restrict__ hb, const __bf16* __restrict__ c10t,
    const __bf16* __restrict__ c11t, const float* __restrict__ PGB,
    const float* __restrict__ Q, const float* __restrict__ DED,
    const float* __restrict__ mask, __bf16* __restrict__ xb_next,
    float* __restrict__ fout, int last) {
  const int tid = threadIdx.x, lane = tid & 31, wv = tid >> 5;
  const int n = blockIdx.x / Nn, i = blockIdx.x % Nn;
  const int j0 = wv * 16, cn = lane & 15, chi = lane >> 4;

  // B-frags: h[n,i,j,:] (contiguous rows) and h[n,j,i,:] (rows strided N*C)
  const __bf16* h1row = hb + ((size_t)(n * Nn + i) * Nn + j0) * Cc;
  const __bf16* h2row = hb + ((size_t)(n * Nn + j0) * Nn + i) * Cc;
  const v16bf b10 = frag_rm(h1row, Cc, 0, lane);
  const v16bf b11 = frag_rm(h1row, Cc, 32, lane);
  const v16bf b20 = frag_rm(h2row, Nn * Cc, 0, lane);
  const v16bf b21 = frag_rm(h2row, Nn * Cc, 32, lane);

  const int j = j0 + cn;
  const float mk = mask[(size_t)(n * Nn + i) * Nn + j];
  const bool isdiag = (j == i);

#pragma unroll
  for (int t = 0; t < 4; ++t) {
    v8f acc = {};
    acc = wmma_bf(frag_rm(c10t + t * 16 * Cc, Cc, 0, lane), b10, acc);
    acc = wmma_bf(frag_rm(c10t + t * 16 * Cc, Cc, 32, lane), b11, acc);
    acc = wmma_bf(frag_rm(c11t + t * 16 * Cc, Cc, 0, lane), b20, acc);
    acc = wmma_bf(frag_rm(c11t + t * 16 * Cc, Cc, 32, lane), b21, acc);

    const int s0 = t * 16 + chi * 8;
    float p8[8], q8[8], d8[8];
    load8f(PGB + ((size_t)(n * Nn) + i) * Cc + s0, p8);
    load8f(Q   + ((size_t)(n * Nn) + j) * Cc + s0, q8);
    load8f(DED + ((size_t)(n * Nn) + i) * Cc + s0, d8);

    const size_t o = ((size_t)(n * Nn + i) * Nn + j) * Cc + s0;
    if (last) {
      float4 o0, o1; float vv[8];
#pragma unroll
      for (int r = 0; r < 8; ++r) {
        float v = acc[r] + p8[r] + q8[r];
        if (isdiag) v += d8[r];
        vv[r] = leakyf(v) * mk;
      }
      o0.x = vv[0]; o0.y = vv[1]; o0.z = vv[2]; o0.w = vv[3];
      o1.x = vv[4]; o1.y = vv[5]; o1.z = vv[6]; o1.w = vv[7];
      *(float4*)(fout + o) = o0;
      *(float4*)(fout + o + 4) = o1;
    } else {
      v8bf ob;
#pragma unroll
      for (int r = 0; r < 8; ++r) {
        float v = acc[r] + p8[r] + q8[r];
        if (isdiag) v += d8[r];
        ob[r] = (__bf16)(leakyf(v) * mk);
      }
      *(v8bf*)(xb_next + o) = ob;
    }
  }
}

extern "C" void kernel_launch(void* const* d_in, const int* in_sizes, int n_in,
                              void* d_out, int out_size, void* d_ws, size_t ws_size,
                              hipStream_t stream) {
  const float* x     = (const float*)d_in[0];
  const float* mask  = (const float*)d_in[1];
  const float* w1    = (const float*)d_in[2];
  const float* b1    = (const float*)d_in[3];
  const float* w2    = (const float*)d_in[4];
  const float* b2    = (const float*)d_in[5];
  const float* w3    = (const float*)d_in[6];
  const float* b3    = (const float*)d_in[7];
  const float* coefs = (const float*)d_in[8];
  const float* bias  = (const float*)d_in[9];
  const float* dbias = (const float*)d_in[10];
  (void)in_sizes; (void)n_in; (void)out_size; (void)ws_size;

  char* ws = (char*)d_ws;
  size_t off = 0;
  auto carve = [&](size_t bytes) -> char* {
    char* p = ws + off;
    off = (off + bytes + 255) & ~(size_t)255;
    return p;
  };
  __bf16* xb     = (__bf16*)carve((size_t)MP * Cc * sizeof(__bf16));
  __bf16* hb     = (__bf16*)carve((size_t)MP * Cc * sizeof(__bf16));
  __bf16* w1b    = (__bf16*)carve((size_t)Ll * Hh * Cc * sizeof(__bf16));
  __bf16* w2b    = (__bf16*)carve((size_t)Ll * Hh * Hh * sizeof(__bf16));
  __bf16* w3b    = (__bf16*)carve((size_t)Ll * Cc * Hh * sizeof(__bf16));
  __bf16* c10t   = (__bf16*)carve((size_t)Ll * Cc * Cc * sizeof(__bf16));
  __bf16* c11t   = (__bf16*)carve((size_t)Ll * Cc * Cc * sizeof(__bf16));
  float* rowS    = (float*)carve((size_t)Bn * Cc * Nn * sizeof(float));
  float* colS    = (float*)carve((size_t)Bn * Cc * Nn * sizeof(float));
  float* diagv   = (float*)carve((size_t)Bn * Cc * Nn * sizeof(float));
  float* sumdiag = (float*)carve((size_t)Bn * Cc * sizeof(float));
  float* sumall  = (float*)carve((size_t)Bn * Cc * sizeof(float));
  float* PGBb    = (float*)carve((size_t)Bn * Nn * Cc * sizeof(float));
  float* Qb      = (float*)carve((size_t)Bn * Nn * Cc * sizeof(float));
  float* DEDb    = (float*)carve((size_t)Bn * Nn * Cc * sizeof(float));
  float* Gb      = (float*)carve((size_t)Bn * Cc * sizeof(float));
  float* Eb      = (float*)carve((size_t)Bn * Cc * sizeof(float));

  // one-time converts: activations + weights -> bf16, coef slices -> bf16^T
  cvt_kernel<<<2048, 256, 0, stream>>>(x, xb, MP * Cc);
  cvt_kernel<<<96, 256, 0, stream>>>(w1, w1b, Ll * Hh * Cc);
  cvt_kernel<<<192, 256, 0, stream>>>(w2, w2b, Ll * Hh * Hh);
  cvt_kernel<<<96, 256, 0, stream>>>(w3, w3b, Ll * Cc * Hh);
  coefT_kernel<<<(Ll * Cc * Cc + 255) / 256, 256, 0, stream>>>(coefs, c10t, c11t);

  for (int l = 0; l < Ll; ++l) {
    const float* cf = coefs + (size_t)l * Cc * Cc * 15;
    msgnet_kernel<<<Bn * Nn, 256, 0, stream>>>(
        xb, mask, w1b + (size_t)l * Hh * Cc, b1 + (size_t)l * Hh,
        w2b + (size_t)l * Hh * Hh, b2 + (size_t)l * Hh,
        w3b + (size_t)l * Cc * Hh, b3 + (size_t)l * Cc, hb);
    reduce_kernel<<<Bn * Nn, 64, 0, stream>>>(hb, rowS, colS, diagv);
    reduce2_kernel<<<Bn, 64, 0, stream>>>(diagv, rowS, sumdiag, sumall);
    ge_kernel<<<Bn, 64, 0, stream>>>(cf, sumdiag, sumall, Gb, Eb);
    rowvec_kernel<<<Bn * Nn, 64, 0, stream>>>(cf, rowS, colS, diagv, Gb, Eb,
                                              bias + (size_t)l * Cc,
                                              dbias + (size_t)l * Cc,
                                              PGBb, Qb, DEDb);
    const int last = (l == Ll - 1);
    out_kernel<<<Bn * Nn, 256, 0, stream>>>(
        hb, c10t + (size_t)l * Cc * Cc, c11t + (size_t)l * Cc * Cc,
        PGBb, Qb, DEDb, mask, xb, (float*)d_out, last);
  }
}
